// GNNModel_90709709291775
// MI455X (gfx1250) — matrix operand order repaired and checked by reading, not compiled, not run on hardware
//
#include <hip/hip_runtime.h>

typedef __attribute__((ext_vector_type(16))) _Float16 v16h;
typedef __attribute__((ext_vector_type(8)))  _Float16 v8h;
typedef __attribute__((ext_vector_type(8)))  float    v8f;
typedef __attribute__((ext_vector_type(4)))  unsigned int v4u;
typedef __attribute__((ext_vector_type(4)))  int      v4i;
typedef __attribute__((ext_vector_type(8)))  int      v8i_t;

#define HID 64
#define TPB 256

// ---------------- utility kernels ----------------

__global__ void k_fill(float* __restrict__ p, float v, long n) {
  long i = (long)blockIdx.x * blockDim.x + threadIdx.x;
  if (i < n) p[i] = v;
}

__global__ void k_f32_to_f16(const float* __restrict__ s, _Float16* __restrict__ d, long n) {
  long i = (long)blockIdx.x * blockDim.x + threadIdx.x;
  if (i < n) d[i] = (_Float16)s[i];
}

// ---------------- self loops (fill_value='mean') ----------------

__global__ void k_selfloop_accum(const int* __restrict__ ei, const float* __restrict__ ea,
                                 float* __restrict__ cnt, float* __restrict__ loop, int E) {
  int t = blockIdx.x * blockDim.x + threadIdx.x;
  if (t >= E) return;
  int d = ei[E + t];
  atomicAdd(&cnt[d], 1.0f);
#pragma unroll
  for (int j = 0; j < 6; ++j)
    atomicAdd(&loop[(long)d * 6 + j], ea[(long)t * 6 + j]);
}

__global__ void k_selfloop_div(float* __restrict__ loop, const float* __restrict__ cnt, long n6) {
  long i = (long)blockIdx.x * blockDim.x + threadIdx.x;
  if (i >= n6) return;
  float c = cnt[i / 6];
  loop[i] = loop[i] / fmaxf(c, 1.0f);
}

// ---------------- layer-1 node transform (din = 6, too small for WMMA) ----------------

__global__ void k_xform_small(const float* __restrict__ x,
                              const float* __restrict__ wl, const float* __restrict__ bl,
                              const float* __restrict__ wr, const float* __restrict__ br,
                              float* __restrict__ xl, float* __restrict__ xr, long n64) {
  long i = (long)blockIdx.x * blockDim.x + threadIdx.x;
  if (i >= n64) return;
  long n = i >> 6;
  int  k = (int)(i & 63);
  float al = bl[k], ar = br[k];
#pragma unroll
  for (int j = 0; j < 6; ++j) {
    float xv = x[n * 6 + j];
    al += xv * wl[j * HID + k];
    ar += xv * wr[j * HID + k];
  }
  xl[i] = al;
  xr[i] = ar;
}

// ---------------- WMMA node transform: out[nrows x NCOLS] = A[nrows x 64](f16) @ B[64 x NCOLS](f16) + bias ------
// B (weight matrix, <= 8KB f16) is staged to LDS ONCE per workgroup by the Tensor Data Mover
// (tensor_load_to_lds, TENSORcnt); all 8 waves feed WMMA B-fragments from LDS via ds_load.
// Compile-time NCOLS/K -> fully unrolled; NT = NCOLS/16 independent C chains hide the f16 WMMA->WMMA hazard.

template <int NCOLS>
__global__ void k_xform_wmma(const _Float16* __restrict__ A, const _Float16* __restrict__ B,
                             const float* __restrict__ bias, float* __restrict__ out,
                             int nrows) {
  constexpr int K  = HID;          // reduction dim
  constexpr int NT = NCOLS / 16;   // 16-wide column tiles
  __shared__ alignas(32) _Float16 Bs[K * NCOLS];   // only LDS allocation -> offset 0

#if __has_builtin(__builtin_amdgcn_tensor_load_to_lds)
  if (threadIdx.x < 32) {                          // wave 0 issues the TDM (EXEC ignored by TDM)
    unsigned long long ga = (unsigned long long)B;
    // D# group0: count=1 (valid), lds_addr=0, global_addr[56:0], type=2 ("image")
    v4u g0 = { 1u, 0u, (unsigned)ga,
               (unsigned)((ga >> 32) & 0x01ffffffu) | (2u << 30) };
    // D# group1: data_size=1 (2B); tensor_dim0=NCOLS, tensor_dim1=K;
    //            tile_dim0=NCOLS, tile_dim1=K (full 2D tile); dim0_stride=NCOLS
    v8i_t g1 = { 0x10000,
                 (int)(((unsigned)NCOLS & 0xffffu) << 16),   // tensor_dim0[15:0]
                 (int)(((unsigned)K     & 0xffffu) << 16),   // tensor_dim1[15:0]
                 (int)(((unsigned)NCOLS & 0xffffu) << 16),   // tile_dim0
                 K,                                          // tile_dim1 (tile_dim2=0)
                 NCOLS,                                      // tensor_dim0_stride[31:0]
                 0, 0 };
    v4i gz = { 0, 0, 0, 0 };
#if __clang_major__ >= 23
    v8i_t gz8 = { 0, 0, 0, 0, 0, 0, 0, 0 };
    __builtin_amdgcn_tensor_load_to_lds(g0, g1, gz, gz, gz8, 0);
#else
    __builtin_amdgcn_tensor_load_to_lds(g0, g1, gz, gz, 0);
#endif
    __builtin_amdgcn_s_wait_tensorcnt(0);          // s_wait_tensorcnt 0x0
  }
  __syncthreads();                                 // publish LDS to all waves
  // TDM wrote Bs behind the compiler's back: force it to treat LDS as written
  // so the B-fragment ds_loads below are not folded to undef.
  asm volatile("" ::: "memory");
#else
  for (int i = (int)threadIdx.x; i < (K * NCOLS) / 8; i += (int)blockDim.x)
    ((v8h*)Bs)[i] = ((const v8h*)B)[i];
  __syncthreads();
#endif

  int wave = threadIdx.x >> 5;
  int lane = threadIdx.x & 31;
  int row0 = (blockIdx.x * (blockDim.x >> 5) + wave) << 4;
  if (row0 >= nrows) return;            // wave-uniform: EXEC stays all-ones for WMMA
  int half = lane >> 4;                 // 0: K 0..7 / 16..23   1: K 8..15 / 24..31
  int mrow = row0 + (lane & 15);

  // A fragments for kc=0 and kc=32, reused across all NT column tiles
  union { v16h v; v8h h[2]; } a0, a1;
  const _Float16* ap = A + (long)mrow * K + half * 8;
  a0.h[0] = *(const v8h*)(ap);          // K = half*8 .. +7
  a0.h[1] = *(const v8h*)(ap + 16);     // K = 16 + half*8 .. +7
  a1.h[0] = *(const v8h*)(ap + 32);     // K = 32 + half*8 .. +7
  a1.h[1] = *(const v8h*)(ap + 48);     // K = 48 + half*8 .. +7

  v8f c[NT];
#pragma unroll
  for (int t = 0; t < NT; ++t) c[t] = (v8f){};

  // kc = 0 chunk: NT independent WMMAs (fills WMMA->WMMA hazard slots)
#pragma unroll
  for (int t = 0; t < NT; ++t) {
    v16h b;
#pragma unroll
    for (int s = 0; s < 16; ++s)
      b[s] = Bs[((s < 8 ? 0 : 8) + half * 8 + s) * NCOLS + t * 16 + (lane & 15)];   // kk = 8h+s / 16+8h+(s-8)
    c[t] = __builtin_amdgcn_wmma_f32_16x16x32_f16(false, a0.v, false, b,
                                                  (short)0, c[t], false, false);
  }
  // kc = 32 chunk
#pragma unroll
  for (int t = 0; t < NT; ++t) {
    v16h b;
#pragma unroll
    for (int s = 0; s < 16; ++s)
      b[s] = Bs[(32 + (s < 8 ? 0 : 8) + half * 8 + s) * NCOLS + t * 16 + (lane & 15)];   // kk = 32 + table order
    c[t] = __builtin_amdgcn_wmma_f32_16x16x32_f16(false, a1.v, false, b,
                                                  (short)0, c[t], false, false);
  }

#pragma unroll
  for (int t = 0; t < NT; ++t) {
    int ncol = t * 16 + (lane & 15);
    float bv = bias[ncol];
#pragma unroll
    for (int r = 0; r < 8; ++r) {
      int m = row0 + r + half * 8;
      out[(long)m * NCOLS + ncol] = c[t][r] + bv;
    }
  }
}

// ---------------- GATv2 attention logits: e = lrelu(xl[s]+xr[d]+ea@we) . att ----------------

__global__ void k_attn_logits(const int* __restrict__ ei, const float* __restrict__ ea,
                              const float* __restrict__ loop,
                              const float* __restrict__ xl, const float* __restrict__ xr,
                              const float* __restrict__ we, const float* __restrict__ att,
                              float* __restrict__ ev, float* __restrict__ emax,
                              int E, int EP) {
  __shared__ float we_s[6 * HID];
  __shared__ float att_s[HID];
  for (int i = threadIdx.x; i < 6 * HID; i += blockDim.x) we_s[i] = we[i];
  for (int i = threadIdx.x; i < HID; i += blockDim.x) att_s[i] = att[i];
  __syncthreads();

  int t = blockIdx.x * blockDim.x + threadIdx.x;
  if (t >= EP) return;
  int s, d; const float* eap;
  if (t < E) { s = ei[t]; d = ei[E + t]; eap = ea + (long)t * 6; }
  else       { s = t - E; d = s;         eap = loop + (long)s * 6; }
  float e0 = eap[0], e1 = eap[1], e2 = eap[2], e3 = eap[3], e4 = eap[4], e5 = eap[5];
  const float* xls = xl + (long)s * HID;
  const float* xrd = xr + (long)d * HID;
  float acc = 0.f;
#pragma unroll 8
  for (int k = 0; k < HID; ++k) {
    float xe = e0 * we_s[0 * HID + k] + e1 * we_s[1 * HID + k] + e2 * we_s[2 * HID + k]
             + e3 * we_s[3 * HID + k] + e4 * we_s[4 * HID + k] + e5 * we_s[5 * HID + k];
    float m = xls[k] + xrd[k] + xe;
    m = (m > 0.f) ? m : 0.2f * m;        // leaky_relu 0.2
    acc += m * att_s[k];
  }
  ev[t] = acc;
  atomicMax(&emax[d], acc);              // -> global_atomic_max_num_f32 (L2 resident)
}

// ---------------- softmax: ex = exp(e - emax[d]); den[d] += ex ----------------

__global__ void k_attn_exp(const int* __restrict__ ei, float* __restrict__ ev,
                           const float* __restrict__ emax, float* __restrict__ den,
                           int E, int EP) {
  int t = blockIdx.x * blockDim.x + threadIdx.x;
  if (t >= EP) return;
  int d = (t < E) ? ei[E + t] : (t - E);
  float ex = __expf(ev[t] - emax[d]);
  ev[t] = ex;
  atomicAdd(&den[d], ex);
}

// ---------------- aggregate: acc[d] += alpha * xl[s]  (thread per (edge,k), coalesced atomics) ----------------

__global__ void k_attn_agg(const int* __restrict__ ei, const float* __restrict__ ev,
                           const float* __restrict__ den, const float* __restrict__ xl,
                           float* __restrict__ acc, int E, long total) {
  long g = (long)blockIdx.x * blockDim.x + threadIdx.x;
  if (g >= total) return;
  long t = g >> 6;
  int  k = (int)(g & 63);
  int s, d;
  if (t < E) { s = ei[t]; d = ei[E + t]; }
  else       { s = (int)(t - E); d = s; }
  float alpha = ev[t] / (den[d] + 1e-16f);
  atomicAdd(&acc[(long)d * HID + k], alpha * xl[(long)s * HID + k]);
}

// ---------------- h = relu(acc + bo); write f16 shadow for next WMMA ----------------

__global__ void k_finalize(const float* __restrict__ acc, const float* __restrict__ bo,
                           _Float16* __restrict__ hf, long n64) {
  long i = (long)blockIdx.x * blockDim.x + threadIdx.x;
  if (i >= n64) return;
  float v = acc[i] + bo[i & 63];
  v = fmaxf(v, 0.f);
  hf[i] = (_Float16)v;
}

// ---------------- final head: out = relu(m32) @ wout + bout ----------------

__global__ void k_mlp_out(const float* __restrict__ m32, const float* __restrict__ wout,
                          const float* __restrict__ bout, float* __restrict__ out, int N) {
  int n = blockIdx.x * blockDim.x + threadIdx.x;
  if (n >= N) return;
  float o0 = bout[0], o1 = bout[1], o2 = bout[2], o3 = bout[3];
#pragma unroll
  for (int k = 0; k < 32; ++k) {
    float v = fmaxf(m32[(long)n * 32 + k], 0.f);
    o0 += v * wout[k * 4 + 0];
    o1 += v * wout[k * 4 + 1];
    o2 += v * wout[k * 4 + 2];
    o3 += v * wout[k * 4 + 3];
  }
  out[(long)n * 4 + 0] = o0;
  out[(long)n * 4 + 1] = o1;
  out[(long)n * 4 + 2] = o2;
  out[(long)n * 4 + 3] = o3;
}

// ---------------- host ----------------

static inline int gdiv(long n, int b) { return (int)((n + b - 1) / b); }

extern "C" void kernel_launch(void* const* d_in, const int* in_sizes, int n_in,
                              void* d_out, int out_size, void* d_ws, size_t ws_size,
                              hipStream_t stream) {
  const int N  = in_sizes[0] / 6;
  const int E  = in_sizes[1] / 2;
  const int EP = E + N;
  const long N64 = (long)N * HID;

  const float* x  = (const float*)d_in[0];
  const int*   ei = (const int*)d_in[1];
  const float* ea = (const float*)d_in[2];
  const float* WL[3]  = {(const float*)d_in[3],  (const float*)d_in[10], (const float*)d_in[17]};
  const float* BL[3]  = {(const float*)d_in[4],  (const float*)d_in[11], (const float*)d_in[18]};
  const float* WR[3]  = {(const float*)d_in[5],  (const float*)d_in[12], (const float*)d_in[19]};
  const float* BR[3]  = {(const float*)d_in[6],  (const float*)d_in[13], (const float*)d_in[20]};
  const float* WE[3]  = {(const float*)d_in[7],  (const float*)d_in[14], (const float*)d_in[21]};
  const float* ATT[3] = {(const float*)d_in[8],  (const float*)d_in[15], (const float*)d_in[22]};
  const float* BO[3]  = {(const float*)d_in[9],  (const float*)d_in[16], (const float*)d_in[23]};
  const float* wlin = (const float*)d_in[24];
  const float* blin = (const float*)d_in[25];
  const float* wout = (const float*)d_in[26];
  const float* bout = (const float*)d_in[27];
  float* out = (float*)d_out;

  // workspace carve-out (256B aligned)
  char* w = (char*)d_ws;
  size_t off = 0;
  auto carve = [&](size_t bytes) -> void* {
    void* p = w + off;
    off = (off + bytes + 255) & ~(size_t)255;
    return p;
  };
  float* cnt   = (float*)carve(sizeof(float) * N);
  float* loop  = (float*)carve(sizeof(float) * (size_t)N * 6);
  float* xl    = (float*)carve(sizeof(float) * N64);
  float* xr    = (float*)carve(sizeof(float) * N64);
  float* ev    = (float*)carve(sizeof(float) * EP);
  float* emax  = (float*)carve(sizeof(float) * N);
  float* den   = (float*)carve(sizeof(float) * N);
  float* acc   = (float*)carve(sizeof(float) * N64);
  float* m32   = (float*)carve(sizeof(float) * (size_t)N * 32);
  _Float16* hf    = (_Float16*)carve(sizeof(_Float16) * N64);
  _Float16* w16l2 = (_Float16*)carve(sizeof(_Float16) * HID * HID);
  _Float16* w16r2 = (_Float16*)carve(sizeof(_Float16) * HID * HID);
  _Float16* w16l3 = (_Float16*)carve(sizeof(_Float16) * HID * HID);
  _Float16* w16r3 = (_Float16*)carve(sizeof(_Float16) * HID * HID);
  _Float16* w16lin = (_Float16*)carve(sizeof(_Float16) * HID * 32);
  (void)ws_size; (void)n_in; (void)out_size;

  // weights -> f16 (once per launch; deterministic)
  k_f32_to_f16<<<gdiv(HID * HID, TPB), TPB, 0, stream>>>(WL[1], w16l2, HID * HID);
  k_f32_to_f16<<<gdiv(HID * HID, TPB), TPB, 0, stream>>>(WR[1], w16r2, HID * HID);
  k_f32_to_f16<<<gdiv(HID * HID, TPB), TPB, 0, stream>>>(WL[2], w16l3, HID * HID);
  k_f32_to_f16<<<gdiv(HID * HID, TPB), TPB, 0, stream>>>(WR[2], w16r3, HID * HID);
  k_f32_to_f16<<<gdiv(HID * 32, TPB), TPB, 0, stream>>>(wlin, w16lin, HID * 32);

  // self loops with mean edge_attr
  k_fill<<<gdiv(N, TPB), TPB, 0, stream>>>(cnt, 0.f, N);
  k_fill<<<gdiv((long)N * 6, TPB), TPB, 0, stream>>>(loop, 0.f, (long)N * 6);
  k_selfloop_accum<<<gdiv(E, TPB), TPB, 0, stream>>>(ei, ea, cnt, loop, E);
  k_selfloop_div<<<gdiv((long)N * 6, TPB), TPB, 0, stream>>>(loop, cnt, (long)N * 6);

  const _Float16* wl16[3] = {nullptr, w16l2, w16l3};
  const _Float16* wr16[3] = {nullptr, w16r2, w16r3};

  for (int l = 0; l < 3; ++l) {
    // node transforms
    if (l == 0) {
      k_xform_small<<<gdiv(N64, TPB), TPB, 0, stream>>>(x, WL[0], BL[0], WR[0], BR[0], xl, xr, N64);
    } else {
      int blocks = gdiv(N, 16 * (TPB / 32));
      k_xform_wmma<HID><<<blocks, TPB, 0, stream>>>(hf, wl16[l], BL[l], xl, N);
      k_xform_wmma<HID><<<blocks, TPB, 0, stream>>>(hf, wr16[l], BR[l], xr, N);
    }
    // per-layer reductions init
    k_fill<<<gdiv(N, TPB), TPB, 0, stream>>>(emax, -1e30f, N);
    k_fill<<<gdiv(N, TPB), TPB, 0, stream>>>(den, 0.f, N);
    k_fill<<<gdiv(N64, TPB), TPB, 0, stream>>>(acc, 0.f, N64);
    // attention + softmax + aggregate
    k_attn_logits<<<gdiv(EP, TPB), TPB, 0, stream>>>(ei, ea, loop, xl, xr, WE[l], ATT[l], ev, emax, E, EP);
    k_attn_exp<<<gdiv(EP, TPB), TPB, 0, stream>>>(ei, ev, emax, den, E, EP);
    k_attn_agg<<<gdiv((long)EP * HID, TPB), TPB, 0, stream>>>(ei, ev, den, xl, acc, E, (long)EP * HID);
    // relu(+bias) -> f16 activations for next dense stage
    k_finalize<<<gdiv(N64, TPB), TPB, 0, stream>>>(acc, BO[l], hf, N64);
  }

  // head: m32 = hf @ wlin + blin (WMMA), then out = relu(m32) @ wout + bout
  int blocks = gdiv(N, 16 * (TPB / 32));
  k_xform_wmma<32><<<blocks, TPB, 0, stream>>>(hf, w16lin, blin, m32, N);
  k_mlp_out<<<gdiv(N, TPB), TPB, 0, stream>>>(m32, wout, bout, out, N);
}
